// GNN_25383256719866
// MI455X (gfx1250) — matrix operand order repaired and checked by reading, not compiled, not run on hardware
//
#include <hip/hip_runtime.h>
#include <stdint.h>

typedef __attribute__((ext_vector_type(16))) __bf16 v16bf;
typedef __attribute__((ext_vector_type(8)))  float  v8f;

#define NN   12500           // nodes
#define NE   50000           // edges
#define CIN  47              // in/out channels
#define CP   48              // padded channels
#define HD   39              // edge-MLP hidden
#define ZP   40              // z padded (39 + 1.0 sentinel for b2 term)
#define ED   15              // edge_attr dim
#define LD   154             // head hidden dim
#define NG   512             // graphs
#define KTOT 1888            // 47*40=1880 padded to 59*32
#define KST  59              // K steps of 32

// ---------- small utility kernels ----------

__global__ void k_pad_x(const float* __restrict__ x, float* __restrict__ xp) {
  int idx = blockIdx.x * blockDim.x + threadIdx.x;
  if (idx >= NN * CP) return;
  int n = idx / CP, c = idx - n * CP;
  xp[idx] = (c < CIN) ? x[n * CIN + c] : 0.f;
}

__global__ void k_zero(float* __restrict__ p, int n) {
  int i = blockIdx.x * blockDim.x + threadIdx.x;
  if (i < n) p[i] = 0.f;
}

// Pre-permute W2 [39,2209] (+b2) into K-major bf16 B-matrix W2t[o][k], k = i*40+h.
// h<39 -> W2[h, i*47+o]; h==39 -> b2[i*47+o] (pairs with the z'==1 sentinel).
__global__ void k_prep_w2(const float* __restrict__ W2, const float* __restrict__ b2,
                          unsigned short* __restrict__ w2t) {
  int idx = blockIdx.x * blockDim.x + threadIdx.x;
  if (idx >= CP * KTOT) return;
  int o = idx / KTOT, k = idx - o * KTOT;
  float v = 0.f;
  if (o < CIN && k < CIN * ZP) {
    int i = k / ZP, h = k - i * ZP;
    if (h < HD)       v = W2[h * (CIN * CIN) + i * CIN + o];
    else if (h == HD) v = b2[i * CIN + o];
  }
  w2t[idx] = __builtin_bit_cast(unsigned short, (__bf16)v);
}

// z' = relu(ea @ W1 + b1), plus sentinel column z'[39] = 1.0
__global__ void k_edge_z(const float* __restrict__ ea, const float* __restrict__ W1,
                         const float* __restrict__ b1, float* __restrict__ z) {
  __shared__ float sW1[ED * HD];
  __shared__ float sb1[HD];
  for (int i = threadIdx.x; i < ED * HD; i += blockDim.x) sW1[i] = W1[i];
  for (int i = threadIdx.x; i < HD; i += blockDim.x) sb1[i] = b1[i];
  __syncthreads();
  int e = blockIdx.x * blockDim.x + threadIdx.x;
  if (e >= NE) return;
  float a[ED];
#pragma unroll
  for (int k = 0; k < ED; ++k) a[k] = ea[e * ED + k];
#pragma unroll 1
  for (int h = 0; h < HD; ++h) {
    float acc = sb1[h];
#pragma unroll
    for (int k = 0; k < ED; ++k) acc += a[k] * sW1[k * HD + h];
    z[e * ZP + h] = acc > 0.f ? acc : 0.f;
  }
  z[e * ZP + HD] = 1.0f;
}

// ---------- fused edge-conditioned message kernel (WMMA) ----------
// One wave per 16-edge tile: msg_tile[16,48] = A'[16,1888] @ W2t^T, A' built on the
// fly from LDS-staged x_src rows and z rows; scatter-add into agg with f32 atomics.
__global__ void __launch_bounds__(32)
k_edge_msg(const float* __restrict__ xp, const float* __restrict__ z,
           const unsigned short* __restrict__ w2t,
           const int* __restrict__ src, const int* __restrict__ dst,
           float* __restrict__ agg) {
  __shared__ float xs[16 * CP];
  __shared__ float zs[16 * ZP];
  __shared__ int   sdst[16];
  const int lane = threadIdx.x;
  const int e0 = blockIdx.x * 16;

  if (lane < 16) sdst[lane] = dst[e0 + lane];
  for (int i = lane; i < 16 * CP; i += 32) {
    int r = i / CP, c = i - r * CP;
    xs[i] = xp[src[e0 + r] * CP + c];            // padded col 47 is 0
  }
  for (int i = lane; i < 16 * ZP; i += 32) {
    int r = i / ZP, c = i - r * ZP;
    zs[i] = z[(e0 + r) * ZP + c];
  }
  __syncthreads();

  const int m = lane & 15;
  const int abase = (lane < 16) ? 0 : 16;       // A K base per half-wave (this code's contiguous k = 16h + s)
  const int bbase = (lane < 16) ? 0 : 16;       // B K base per half-wave (same contiguous k = 16h + s as A)
  const float* xr = &xs[m * CP];
  const float* zr = &zs[m * ZP];
  const unsigned short* b0p = w2t + (size_t)(m +  0) * KTOT + bbase;
  const unsigned short* b1p = w2t + (size_t)(m + 16) * KTOT + bbase;
  const unsigned short* b2p = w2t + (size_t)(m + 32) * KTOT + bbase;

  v8f acc0 = {0.f,0.f,0.f,0.f,0.f,0.f,0.f,0.f};
  v8f acc1 = acc0, acc2 = acc0;

#pragma unroll 1
  for (int ks = 0; ks < KST; ++ks) {
    const int k0 = ks * 32;
    // A fragment: elements 0..15 -> K = k0+abase+0..15
    v16bf a;
    int kk = k0 + abase;
    int i = kk / ZP, h = kk - i * ZP;
#pragma unroll
    for (int j = 0; j < 8; ++j) {
      a[j] = (__bf16)(xr[i] * zr[h]);
      if (++h == ZP) { h = 0; ++i; }
    }

#pragma unroll
    for (int j = 8; j < 16; ++j) {
      a[j] = (__bf16)(xr[i] * zr[h]);
      if (++h == ZP) { h = 0; ++i; }
    }
    // B fragments: lane = column n, 16 contiguous K per half-wave (32B aligned loads)
    v16bf bf0 = *reinterpret_cast<const v16bf*>(b0p + k0);
    v16bf bf1 = *reinterpret_cast<const v16bf*>(b1p + k0);
    v16bf bf2 = *reinterpret_cast<const v16bf*>(b2p + k0);

    acc0 = __builtin_amdgcn_wmma_f32_16x16x32_bf16(false, a, false, bf0, (short)0, acc0, false, false);
    acc1 = __builtin_amdgcn_wmma_f32_16x16x32_bf16(false, a, false, bf1, (short)0, acc1, false, false);
    acc2 = __builtin_amdgcn_wmma_f32_16x16x32_bf16(false, a, false, bf2, (short)0, acc2, false, false);
  }

  // D layout: VGPR r -> M = r + (lane<16 ? 0 : 8), N = (lane&15) + 16*ntile
  const int mrow = (lane < 16) ? 0 : 8;
  const int n0 = lane & 15;
#pragma unroll
  for (int r = 0; r < 8; ++r) {
    float* row = agg + (size_t)sdst[r + mrow] * CP;
    atomicAdd(row + n0,      acc0[r]);
    atomicAdd(row + n0 + 16, acc1[r]);
    if (n0 + 32 < CIN) atomicAdd(row + n0 + 32, acc2[r]);
  }
}

// out[n,o] = agg[n,o] + x[n,:] @ root[:,o] + bias[o]  (padded col 47 = 0)
__global__ void k_node_finish(const float* __restrict__ xp, const float* __restrict__ agg,
                              const float* __restrict__ root, const float* __restrict__ bias,
                              float* __restrict__ outp) {
  int idx = blockIdx.x * blockDim.x + threadIdx.x;
  if (idx >= NN * CP) return;
  int n = idx / CP, o = idx - n * CP;
  float v = 0.f;
  if (o < CIN) {
    v = agg[idx] + bias[o];
    const float* xrow = xp + n * CP;
#pragma unroll 1
    for (int i = 0; i < CIN; ++i) v += xrow[i] * root[i * CIN + o];
  }
  outp[idx] = v;
}

__global__ void k_pool(const float* __restrict__ outp, const int* __restrict__ batch,
                       float* __restrict__ g) {
  int idx = blockIdx.x * blockDim.x + threadIdx.x;
  if (idx >= NN * CP) return;
  int n = idx / CP, o = idx - n * CP;
  if (o < CIN) atomicAdd(&g[batch[n] * CP + o], outp[idx]);
}

// Fused readout: lin0 + 6 residual MLP blocks + both heads, one workgroup per graph.
__global__ void __launch_bounds__(192)
k_head(const float* __restrict__ g,
       const float* __restrict__ lin0W, const float* __restrict__ lin0b,
       const float* __restrict__ rW1, const float* __restrict__ rb1,
       const float* __restrict__ rW2, const float* __restrict__ rb2,
       const float* __restrict__ sizeW, const float* __restrict__ sizeb,
       const float* __restrict__ rseW, const float* __restrict__ rseb,
       float* __restrict__ out) {
  __shared__ float h[LD];
  __shared__ float t[LD];
  const int gi = blockIdx.x;
  const int c = threadIdx.x;
  if (c < LD) {
    float acc = lin0b[c];
    const float* grow = g + gi * CP;
#pragma unroll 1
    for (int i = 0; i < CIN; ++i) acc += grow[i] * lin0W[i * LD + c];
    h[c] = acc;
  }
  __syncthreads();
  for (int l = 0; l < 6; ++l) {
    if (c < LD) {
      float acc = rb1[l * LD + c];
      const float* W = rW1 + (size_t)l * LD * LD;
#pragma unroll 1
      for (int j = 0; j < LD; ++j) acc += h[j] * W[j * LD + c];
      t[c] = acc > 0.f ? acc : 0.f;
    }
    __syncthreads();
    float hn = 0.f;
    if (c < LD) {
      float acc = rb2[l * LD + c] + h[c];
      const float* W = rW2 + (size_t)l * LD * LD;
#pragma unroll 1
      for (int j = 0; j < LD; ++j) acc += t[j] * W[j * LD + c];
      hn = acc;
    }
    __syncthreads();
    if (c < LD) h[c] = hn;
    __syncthreads();
  }
  if (c < 7) {
    float acc = sizeb[c];
#pragma unroll 1
    for (int j = 0; j < LD; ++j) acc += h[j] * sizeW[j * 7 + c];
    out[gi * 7 + c] = acc;
  } else if (c == 7) {
    float acc = rseb[0];
#pragma unroll 1
    for (int j = 0; j < LD; ++j) acc += h[j] * rseW[j];
    out[NG * 7 + gi] = acc;
  }
}

extern "C" void kernel_launch(void* const* d_in, const int* in_sizes, int n_in,
                              void* d_out, int out_size, void* d_ws, size_t ws_size,
                              hipStream_t stream) {
  (void)in_sizes; (void)n_in; (void)out_size; (void)ws_size;
  const float* x    = (const float*)d_in[0];
  const int*   ei   = (const int*)d_in[1];
  const float* ea   = (const float*)d_in[2];
  const int*   batch= (const int*)d_in[3];
  const float* c1W1 = (const float*)d_in[4],  *c1b1 = (const float*)d_in[5];
  const float* c1W2 = (const float*)d_in[6],  *c1b2 = (const float*)d_in[7];
  const float* c2W1 = (const float*)d_in[8],  *c2b1 = (const float*)d_in[9];
  const float* c2W2 = (const float*)d_in[10], *c2b2 = (const float*)d_in[11];
  const float* c1root = (const float*)d_in[12], *c1bias = (const float*)d_in[13];
  const float* c2root = (const float*)d_in[14], *c2bias = (const float*)d_in[15];
  const float* lin0W  = (const float*)d_in[16], *lin0b  = (const float*)d_in[17];
  const float* rW1 = (const float*)d_in[18], *rb1 = (const float*)d_in[19];
  const float* rW2 = (const float*)d_in[20], *rb2 = (const float*)d_in[21];
  const float* sW  = (const float*)d_in[22], *sb  = (const float*)d_in[23];
  const float* rseW= (const float*)d_in[24], *rseb= (const float*)d_in[25];

  char* ws = (char*)d_ws;
  size_t off = 0;
  auto alloc = [&](size_t bytes) { size_t o = off; off = (off + bytes + 255) & ~(size_t)255; return o; };
  float* xp   = (float*)(ws + alloc((size_t)NN * CP * 4));
  float* o1   = (float*)(ws + alloc((size_t)NN * CP * 4));
  float* o2   = (float*)(ws + alloc((size_t)NN * CP * 4));
  float* agg  = (float*)(ws + alloc((size_t)NN * CP * 4));
  float* zbuf = (float*)(ws + alloc((size_t)NE * ZP * 4));
  unsigned short* w2t = (unsigned short*)(ws + alloc((size_t)CP * KTOT * 2));
  float* gbuf = (float*)(ws + alloc((size_t)NG * CP * 4));

  const int* src = ei;
  const int* dst = ei + NE;
  const int T = 256;
  const int gNC = (NN * CP + T - 1) / T;

  // ---- layer 1 ----
  k_pad_x   <<<gNC, T, 0, stream>>>(x, xp);
  k_zero    <<<gNC, T, 0, stream>>>(agg, NN * CP);
  k_prep_w2 <<<(CP * KTOT + T - 1) / T, T, 0, stream>>>(c1W2, c1b2, w2t);
  k_edge_z  <<<(NE + T - 1) / T, T, 0, stream>>>(ea, c1W1, c1b1, zbuf);
  k_edge_msg<<<NE / 16, 32, 0, stream>>>(xp, zbuf, w2t, src, dst, agg);
  k_node_finish<<<gNC, T, 0, stream>>>(xp, agg, c1root, c1bias, o1);
  // ---- layer 2 ----
  k_zero    <<<gNC, T, 0, stream>>>(agg, NN * CP);
  k_prep_w2 <<<(CP * KTOT + T - 1) / T, T, 0, stream>>>(c2W2, c2b2, w2t);
  k_edge_z  <<<(NE + T - 1) / T, T, 0, stream>>>(ea, c2W1, c2b1, zbuf);
  k_edge_msg<<<NE / 16, 32, 0, stream>>>(o1, zbuf, w2t, src, dst, agg);
  k_node_finish<<<gNC, T, 0, stream>>>(o1, agg, c2root, c2bias, o2);
  // ---- readout ----
  k_zero<<<(NG * CP + T - 1) / T, T, 0, stream>>>(gbuf, NG * CP);
  k_pool<<<gNC, T, 0, stream>>>(o2, batch, gbuf);
  k_head<<<NG, 192, 0, stream>>>(gbuf, lin0W, lin0b, rW1, rb1, rW2, rb2,
                                 sW, sb, rseW, rseb, (float*)d_out);
}
